// NeighborhoodSearch_48971217109027
// MI455X (gfx1250) — matrix operand order repaired and checked by reading, not compiled, not run on hardware
//
#include <hip/hip_runtime.h>
#include <hip/hip_bf16.h>

// ---------------------------------------------------------------------------
// Problem constants (from reference setup_inputs)
// ---------------------------------------------------------------------------
#define BATCH   2
#define HGRID   48
#define NTOK    (HGRID * HGRID)      // 2304
#define DMODEL  768
#define NHEADS  8
#define HDIM    (DMODEL / NHEADS)    // 96
#define LN_EPSF 1e-5f

typedef unsigned short ushort_t;
typedef unsigned int   uint_t;

// ---------------------------------------------------------------------------
// WMMA types / helpers (CDNA5 wave32, v_wmma_f32_16x16x32_bf16)
// ---------------------------------------------------------------------------
typedef __attribute__((ext_vector_type(16))) __bf16 v16bf;
typedef __attribute__((ext_vector_type(8)))  float  v8f;

union BFPack { v16bf v; uint_t u[8]; uint4 q[2]; };

// Truncating f32->bf16 pair pack: one v_perm_b32 (bytes {hi.3,hi.2,lo.3,lo.2}).
__device__ __forceinline__ uint_t packbf(float lo, float hi) {
  return __builtin_amdgcn_perm(__float_as_uint(hi), __float_as_uint(lo),
                               0x07060302u);
}
__device__ __forceinline__ ushort_t f2bfs(float f) {
  return (ushort_t)(__float_as_uint(f) >> 16);
}
__device__ __forceinline__ v8f wmma_bf16(v16bf a, v16bf b, v8f c) {
  // 8 args: (neg_a, A, neg_b, B, c_mod, C, reuse_a, reuse_b)
  return __builtin_amdgcn_wmma_f32_16x16x32_bf16(false, a, false, b, (short)0, c,
                                                 false, false);
}

// A fragment: 16x32 bf16 from bf16 row-major memory (leading dim ld, elements).
// lane: m = lane&15, half = lane>>4; reg j holds K = 16*(j/4)+8*half+2*(j%4)+{0,1}
// -> two contiguous 8-bf16 runs per lane => 2x global_load_b128, zero VALU.
__device__ __forceinline__ v16bf load_A_bf(const ushort_t* __restrict__ A, int ld,
                                           int row0, int k0, int lane) {
  const int m = lane & 15, half = lane >> 4;
  const ushort_t* p = A + (size_t)(row0 + m) * ld + k0 + (half << 3);
  BFPack r;
  r.q[0] = *(const uint4*)(p);
  r.q[1] = *(const uint4*)(p + 16);
  return r.v;
}

// B fragment (B[k][n] = M[n0+n][k]) from bf16 memory where "row n" has
// contiguous k.  lane: n = lane&15, half = lane>>4; reg j: K = 16*(j/4)+8*half+2*(j%4)+{0,1}
// -> two contiguous 8-bf16 runs per lane => 2x global_load_b128, zero VALU.
__device__ __forceinline__ v16bf load_B_bf(const ushort_t* __restrict__ M, int ld,
                                           int n0, int k0, int lane) {
  const int n = lane & 15, half = lane >> 4;
  const ushort_t* p = M + (size_t)(n0 + n) * ld + k0 + (half << 3);
  BFPack r;
  r.q[0] = *(const uint4*)(p);
  r.q[1] = *(const uint4*)(p + 16);
  return r.v;
}

// ---------------------------------------------------------------------------
// Kernel 0: one-time f32 -> bf16 conversion (weights, x1)
// ---------------------------------------------------------------------------
__global__ void __launch_bounds__(256)
cvt_bf16_kernel(const float* __restrict__ src, ushort_t* __restrict__ dst, int n8) {
  const int i = blockIdx.x * blockDim.x + threadIdx.x;
  if (i >= n8) return;
  const float4 f0 = *(const float4*)(src + (size_t)i * 8);
  const float4 f1 = *(const float4*)(src + (size_t)i * 8 + 4);
  uint4 u;
  u.x = packbf(f0.x, f0.y);  u.y = packbf(f0.z, f0.w);
  u.z = packbf(f1.x, f1.y);  u.w = packbf(f1.z, f1.w);
  *(uint4*)(dst + (size_t)i * 8) = u;
}

// ---------------------------------------------------------------------------
// Kernel 1: neighbor-sum (softmax over singleton axis == uniform weights)
//           + LayerNorm.  One wave per token row of 768.  bf16 output.
// ---------------------------------------------------------------------------
__global__ void __launch_bounds__(256)
fused_ln_kernel(const float* __restrict__ x2, const float* __restrict__ gamma,
                const float* __restrict__ beta, ushort_t* __restrict__ out) {
  const int lane = threadIdx.x & 31;
  const int gw   = (blockIdx.x * (blockDim.x >> 5)) + (threadIdx.x >> 5);
  const int b = gw / NTOK;
  const int n = gw - b * NTOK;

  const int r = n / HGRID, c = n - r * HGRID;
  int off[9];
  int cnt = 0;
#pragma unroll
  for (int dr = -1; dr <= 1; ++dr)
#pragma unroll
    for (int dc = -1; dc <= 1; ++dc) {
      const int rr = r + dr, cc = c + dc;
      if (rr >= 0 && rr < HGRID && cc >= 0 && cc < HGRID)
        off[cnt++] = (rr * HGRID + cc) * DMODEL;
    }
  for (int i = cnt; i < 9; ++i) off[i] = off[0];   // pad with first valid (ref semantics)

  const float* base = x2 + (size_t)b * NTOK * DMODEL;
  float4 vals[DMODEL / 128];
  float s = 0.f, s2 = 0.f;
#pragma unroll
  for (int i = 0; i < DMODEL / 128; ++i) {
    const int d = lane * 4 + 128 * i;
    float4 a = make_float4(0.f, 0.f, 0.f, 0.f);
#pragma unroll
    for (int kk = 0; kk < 9; ++kk) {
      const float4 t = *(const float4*)(base + off[kk] + d);
      a.x += t.x; a.y += t.y; a.z += t.z; a.w += t.w;
    }
    vals[i] = a;
    s  += a.x + a.y + a.z + a.w;
    s2 += a.x * a.x + a.y * a.y + a.z * a.z + a.w * a.w;
  }
#pragma unroll
  for (int m = 16; m >= 1; m >>= 1) {
    s  += __shfl_xor(s,  m, 32);
    s2 += __shfl_xor(s2, m, 32);
  }
  const float mu   = s * (1.0f / DMODEL);
  const float var  = s2 * (1.0f / DMODEL) - mu * mu;
  const float rstd = rsqrtf(var + LN_EPSF);

  ushort_t* orow = out + (size_t)gw * DMODEL;
#pragma unroll
  for (int i = 0; i < DMODEL / 128; ++i) {
    const int d = lane * 4 + 128 * i;
    const float4 g  = *(const float4*)(gamma + d);
    const float4 be = *(const float4*)(beta + d);
    uint2 u;
    u.x = packbf((vals[i].x - mu) * rstd * g.x + be.x,
                 (vals[i].y - mu) * rstd * g.y + be.y);
    u.y = packbf((vals[i].z - mu) * rstd * g.z + be.z,
                 (vals[i].w - mu) * rstd * g.w + be.w);
    *(uint2*)(orow + d) = u;
  }
}

// ---------------------------------------------------------------------------
// Kernel 2: C = (A[M,K]*W[Nout,K]^T + bias) * scale, A/W bf16, 32x64 per wave.
// OUT_MODE 0: f32 row-major; 1: bf16 row-major; 2: bf16 transposed-per-head
//             ([b][col][token], token-contiguous -> b128 packed stores).
// ---------------------------------------------------------------------------
template <int OUT_MODE>
__global__ void __launch_bounds__(256)
gemm_bf16_kernel(const ushort_t* __restrict__ A, const ushort_t* __restrict__ Wb,
                 const float* __restrict__ bias, void* __restrict__ Cout,
                 int K, int Nout, float scale) {
  const int lane = threadIdx.x & 31;
  const int wave = (blockIdx.x * (blockDim.x >> 5)) + (threadIdx.x >> 5);
  const int nstrips = Nout >> 6;
  const int mt = wave / nstrips;
  const int ns = wave - mt * nstrips;
  const int row0 = mt << 5;       // 32-row tile (never crosses a batch: 2304%32==0)
  const int n0   = ns << 6;       // 64-col strip

  v8f acc[2][4];
  const v8f zero = {};
#pragma unroll
  for (int i = 0; i < 2; ++i)
#pragma unroll
    for (int t = 0; t < 4; ++t) acc[i][t] = zero;

  for (int k0 = 0; k0 < K; k0 += 32) {
    if (k0 + 32 < K)  // speculative prefetch of next W panel
      __builtin_prefetch(Wb + (size_t)(n0 + (lane & 15)) * K + k0 + 32, 0, 1);
    const v16bf a0 = load_A_bf(A, K, row0,      k0, lane);
    const v16bf a1 = load_A_bf(A, K, row0 + 16, k0, lane);
#pragma unroll
    for (int t = 0; t < 4; ++t) {
      const v16bf b = load_B_bf(Wb, K, n0 + 16 * t, k0, lane);
      acc[0][t] = wmma_bf16(a0, b, acc[0][t]);
      acc[1][t] = wmma_bf16(a1, b, acc[1][t]);
    }
  }

  const int half = lane >> 4, nl = lane & 15;
  if (OUT_MODE == 2) {
    // transposed per-head bf16: CT[((b*DMODEL + col) * NTOK) + token]
    ushort_t* CT = (ushort_t*)Cout;
    const int bb   = row0 / NTOK;
    const int tokb = row0 - bb * NTOK + (half << 3);
#pragma unroll
    for (int i = 0; i < 2; ++i)
#pragma unroll
      for (int t = 0; t < 4; ++t) {
        const int col = n0 + 16 * t + nl;
        const float bi = bias[col];
        float e[8];
#pragma unroll
        for (int vv = 0; vv < 8; ++vv) e[vv] = (acc[i][t][vv] + bi) * scale;
        uint4 u;
        u.x = packbf(e[0], e[1]);  u.y = packbf(e[2], e[3]);
        u.z = packbf(e[4], e[5]);  u.w = packbf(e[6], e[7]);
        *(uint4*)(CT + ((size_t)bb * DMODEL + col) * NTOK + tokb + 16 * i) = u;
      }
  } else {
#pragma unroll
    for (int i = 0; i < 2; ++i)
#pragma unroll
      for (int vv = 0; vv < 8; ++vv) {
        const int rrow = row0 + 16 * i + vv + (half << 3);
#pragma unroll
        for (int t = 0; t < 4; ++t) {
          const int col = n0 + 16 * t + nl;
          const float val = (acc[i][t][vv] + bias[col]) * scale;
          if (OUT_MODE == 0)
            ((float*)Cout)[(size_t)rrow * Nout + col] = val;
          else
            ((ushort_t*)Cout)[(size_t)rrow * Nout + col] = f2bfs(val);
        }
      }
  }
}

// ---------------------------------------------------------------------------
// Kernel 3: flash attention.  One wave per (b, head, 16-query tile).
// Q/K bf16 row-major, V bf16 transposed-per-head -> all fragments are direct
// b128 loads.  Only P transposes through LDS (C-layout -> A-layout).
// Key tile = 64 (4x16 S tiles per softmax update): amortizes the max
// butterfly, alpha exps and O-rescale over 2x more keys.  Row-sums of P are
// accumulated on the MAC array via an all-ones B column (no sum butterflies).
// ---------------------------------------------------------------------------
#define QT_PER_BH (NTOK / 16)  // 144

__global__ void __launch_bounds__(128)
attention_kernel(const ushort_t* __restrict__ q, const ushort_t* __restrict__ k,
                 const ushort_t* __restrict__ vt, ushort_t* __restrict__ aout) {
  __shared__ ushort_t plds_all[4 * 16 * 64];   // per-wave P tile (16x64 bf16)

  const int lane = threadIdx.x & 31;
  const int warp = threadIdx.x >> 5;
  const int gw   = blockIdx.x * (blockDim.x >> 5) + warp;
  const int qt = gw % QT_PER_BH;
  const int h  = (gw / QT_PER_BH) % NHEADS;
  const int b  = gw / (QT_PER_BH * NHEADS);
  const int half = lane >> 4, nl = lane & 15;

  ushort_t* plds = plds_all + warp * 16 * 64;

  const ushort_t* qb  = q  + ((size_t)b * NTOK + (size_t)qt * 16) * DMODEL + h * HDIM;
  const ushort_t* kb  = k  + (size_t)b * NTOK * DMODEL + h * HDIM;
  const ushort_t* vtb = vt + ((size_t)b * DMODEL + h * HDIM) * NTOK;

  // Q fragments: 16 x 96 = 3 chunks of K=32 (1/sqrt(hd) already folded into q)
  v16bf qa[3];
#pragma unroll
  for (int c = 0; c < 3; ++c) qa[c] = load_A_bf(qb, DMODEL, 0, c * 32, lane);

  // Constant B fragment: column 0 all-ones (bf16 1.0 pair = 0x3F803F80).
  BFPack onesf;
  {
    const uint_t v1 = (nl == 0) ? 0x3F803F80u : 0u;
#pragma unroll
    for (int j = 0; j < 8; ++j) onesf.u[j] = v1;
  }

  v8f o[6], osum;
  const v8f zero = {};
#pragma unroll
  for (int t = 0; t < 6; ++t) o[t] = zero;
  osum = zero;
  float rmax[8];
#pragma unroll
  for (int vv = 0; vv < 8; ++vv) rmax[vv] = -3.0e38f;

  for (int kp = 0; kp < NTOK / 64; ++kp) {
    const int key0 = kp * 64;

    // ---- S = Q * K^T for four 16-key tiles (direct bf16 b128 loads) ----
    v8f s[4];
#pragma unroll
    for (int j = 0; j < 4; ++j) s[j] = zero;
#pragma unroll
    for (int c = 0; c < 3; ++c)
#pragma unroll
      for (int j = 0; j < 4; ++j) {
        const v16bf bk = load_B_bf(kb, DMODEL, key0 + 16 * j, c * 32, lane);
        s[j] = wmma_bf16(qa[c], bk, s[j]);
      }

    // ---- online softmax over the 64 keys; row m = vv + 8*half, col = nl ----
    // Only the running MAX needs a cross-lane butterfly (once per 64 keys);
    // the row-sum is accumulated by the ones-column WMMAs below.
    float alph[8];
#pragma unroll
    for (int vv = 0; vv < 8; ++vv) {
      float mx = fmaxf(fmaxf(s[0][vv], s[1][vv]), fmaxf(s[2][vv], s[3][vv]));
#pragma unroll
      for (int msk = 8; msk >= 1; msk >>= 1) mx = fmaxf(mx, __shfl_xor(mx, msk, 32));
      const float mnew = fmaxf(rmax[vv], mx);
      alph[vv] = __expf(rmax[vv] - mnew);
      rmax[vv] = mnew;
      const int m = vv + (half << 3);
#pragma unroll
      for (int j = 0; j < 4; ++j)
        plds[m * 64 + 16 * j + nl] = f2bfs(__expf(s[j][vv] - mnew));
    }
#pragma unroll
    for (int t = 0; t < 6; ++t)
#pragma unroll
      for (int vv = 0; vv < 8; ++vv) o[t][vv] *= alph[vv];
#pragma unroll
    for (int vv = 0; vv < 8; ++vv) osum[vv] *= alph[vv];

    __syncthreads();   // uniform trip count across all waves in block

    // ---- P (A-layout, two K=32 fragments) from LDS; V^T direct from global --
    BFPack pa0, pa1;
#pragma unroll
    for (int j = 0; j < 8; ++j) {
      const int kk = ((j >> 2) << 4) + (half << 3) + ((j & 3) << 1);
      pa0.u[j] = *(const uint_t*)&plds[nl * 64 + kk];
      pa1.u[j] = *(const uint_t*)&plds[nl * 64 + 32 + kk];
    }
#pragma unroll
    for (int t = 0; t < 6; ++t) {
      const v16bf vf0 = load_B_bf(vtb, NTOK, t * 16, key0,      lane);
      const v16bf vf1 = load_B_bf(vtb, NTOK, t * 16, key0 + 32, lane);
      o[t] = wmma_bf16(pa0.v, vf0, o[t]);
      o[t] = wmma_bf16(pa1.v, vf1, o[t]);
    }
    osum = wmma_bf16(pa0.v, onesf.v, osum);  // row-sum into column 0
    osum = wmma_bf16(pa1.v, onesf.v, osum);
    __syncthreads();   // protect next iteration's P writes
  }

  // ---- normalize (denominator = osum column 0, broadcast within half) ----
#pragma unroll
  for (int vv = 0; vv < 8; ++vv) {
    const float rs  = __shfl(osum[vv], lane & 16, 32);
    const float inv = 1.0f / rs;
    const int m = vv + (half << 3);
    ushort_t* orow =
        aout + ((size_t)b * NTOK + (size_t)qt * 16 + m) * DMODEL + h * HDIM;
#pragma unroll
    for (int t = 0; t < 6; ++t) orow[t * 16 + nl] = f2bfs(o[t][vv] * inv);
  }
}

// ---------------------------------------------------------------------------
// Launch
// ---------------------------------------------------------------------------
extern "C" void kernel_launch(void* const* d_in, const int* in_sizes, int n_in,
                              void* d_out, int out_size, void* d_ws, size_t ws_size,
                              hipStream_t stream) {
  const float* x1    = (const float*)d_in[0];
  const float* x2    = (const float*)d_in[1];
  const float* Wq    = (const float*)d_in[2];
  const float* Wk    = (const float*)d_in[3];
  const float* Wv    = (const float*)d_in[4];
  const float* bq    = (const float*)d_in[5];
  const float* bk    = (const float*)d_in[6];
  const float* bv    = (const float*)d_in[7];
  const float* Wo    = (const float*)d_in[8];
  const float* bo    = (const float*)d_in[9];
  const float* gamma = (const float*)d_in[10];
  const float* beta  = (const float*)d_in[11];
  float* out = (float*)d_out;

  const size_t tok  = (size_t)BATCH * NTOK * DMODEL;  // 3,538,944 elements
  const size_t wsz  = (size_t)DMODEL * DMODEL;        //   589,824 elements
  ushort_t* x1_bf    = (ushort_t*)d_ws;
  ushort_t* fused_bf = x1_bf + tok;
  ushort_t* q_bf     = fused_bf + tok;
  ushort_t* k_bf     = q_bf + tok;
  ushort_t* vt_bf    = k_bf + tok;
  ushort_t* att_bf   = vt_bf + tok;
  ushort_t* wq_bf    = att_bf + tok;
  ushort_t* wk_bf    = wq_bf + wsz;
  ushort_t* wv_bf    = wk_bf + wsz;
  ushort_t* wo_bf    = wv_bf + wsz;

  const int rows = BATCH * NTOK;                      // 4608 (multiple of 32)
  const float qscale = 0.10206207262f;                // HDIM^-0.5 = 1/sqrt(96)

  // (0) one-time f32 -> bf16 converts (weights + x1)
  const int w8 = (int)(wsz / 8), t8 = (int)(tok / 8);
  cvt_bf16_kernel<<<w8 / 256, 256, 0, stream>>>(Wq, wq_bf, w8);
  cvt_bf16_kernel<<<w8 / 256, 256, 0, stream>>>(Wk, wk_bf, w8);
  cvt_bf16_kernel<<<w8 / 256, 256, 0, stream>>>(Wv, wv_bf, w8);
  cvt_bf16_kernel<<<w8 / 256, 256, 0, stream>>>(Wo, wo_bf, w8);
  cvt_bf16_kernel<<<t8 / 256, 256, 0, stream>>>(x1, x1_bf, t8);

  // (1) neighbor-sum + LayerNorm -> bf16: 1 wave/token, 8 waves/block
  fused_ln_kernel<<<rows / 8, 256, 0, stream>>>(x2, gamma, beta, fused_bf);

  // (2) projections: waves = (M/32)*(D/64) = 144*12 = 1728; 8 waves/block
  const int gemm_blocks = (rows / 32) * (DMODEL / 64) / 8;   // 216
  gemm_bf16_kernel<1><<<gemm_blocks, 256, 0, stream>>>(fused_bf, wq_bf, bq, q_bf,
                                                       DMODEL, DMODEL, qscale);
  gemm_bf16_kernel<1><<<gemm_blocks, 256, 0, stream>>>(x1_bf, wk_bf, bk, k_bf,
                                                       DMODEL, DMODEL, 1.0f);
  gemm_bf16_kernel<2><<<gemm_blocks, 256, 0, stream>>>(x1_bf, wv_bf, bv, vt_bf,
                                                       DMODEL, DMODEL, 1.0f);

  // (3) attention: waves = B*H*(N/16) = 2304; 4 waves/block -> 576 blocks
  attention_kernel<<<BATCH * NHEADS * QT_PER_BH / 4, 128, 0, stream>>>(
      q_bf, k_bf, vt_bf, att_bf);

  // (4) output projection -> d_out (f32)
  gemm_bf16_kernel<0><<<gemm_blocks, 256, 0, stream>>>(att_bf, wo_bf, bo, out,
                                                       DMODEL, DMODEL, 1.0f);
}
